// MultiheadAttention_37022618091816
// MI455X (gfx1250) — matrix operand order repaired and checked by reading, not compiled, not run on hardware
//
#include <hip/hip_runtime.h>
#include <hip/hip_bf16.h>
#include <math.h>

// Problem constants (match reference)
#define B_SZ   2
#define S_SZ   2048
#define E_SZ   2048
#define H_Q    16
#define H_KV   4
#define D_HD   128
#define KV_SZ  (H_KV * D_HD)   // 512
#define M_ROWS (B_SZ * S_SZ)   // 4096

typedef __attribute__((ext_vector_type(16))) __bf16 v16bf;
typedef __attribute__((ext_vector_type(8)))  float  v8f;

union FragU { uint4 u[2]; v16bf v; __bf16 b[16]; };

static __device__ inline v8f vzero8() {
  v8f z;
#pragma unroll
  for (int i = 0; i < 8; ++i) z[i] = 0.0f;
  return z;
}

static __device__ inline v8f wmma_bf16(v16bf a, v16bf b, v8f c) {
  // (neg_a, A, neg_b, B, c_mod, C, reuse_a, reuse_b)
  return __builtin_amdgcn_wmma_f32_16x16x32_bf16(false, a, false, b, (short)0, c, false, false);
}

// A-matrix 16x32 bf16 fragment from K-contiguous row-major source.
// lanes 0-15: row m, K = 0..15 ; lanes 16-31: row m, K = 16..31
static __device__ inline v16bf load_afrag(const __bf16* base, int ld, int row, int k, int lane) {
  const __bf16* p = base + (size_t)row * ld + k + ((lane >> 4) << 4);
  FragU f;
  f.u[0] = *(const uint4*)(p);        // 8 bf16 (16 B)
  f.u[1] = *(const uint4*)(p + 8);    // next 8 bf16 along K
  return f.v;
}

// B-matrix 32x16 bf16 fragment; column n of B = row `colrow` of a K-contiguous source.
// lanes 0-15 hold K=0..15, lanes 16-31 hold K=16..31 -> 32 contiguous bytes per lane.
static __device__ inline v16bf load_bfrag(const __bf16* base, int ld, int colrow, int k, int lane) {
  const __bf16* p = base + (size_t)colrow * ld + k + ((lane >> 4) << 4);
  FragU f;
  f.u[0] = *(const uint4*)(p);
  f.u[1] = *(const uint4*)(p + 8);
  return f.v;
}

// ---------------------------------------------------------------------------
// f32 -> bf16 conversion
// ---------------------------------------------------------------------------
__global__ void cvt_bf16_kernel(const float* __restrict__ src, __bf16* __restrict__ dst, long n) {
  long stride = (long)gridDim.x * blockDim.x;
  for (long i = (long)blockIdx.x * blockDim.x + threadIdx.x; i < n; i += stride)
    dst[i] = (__bf16)src[i];
}

// ---------------------------------------------------------------------------
// WMMA GEMM mainloop: out(row,col) = sum_k A[row][k] * W[col][k]
// Per wave: 64 rows x 32 cols = 4x2 wmma accumulators. Direct-from-global
// fragments (whole bf16 working set is L2-resident: 192 MB L2).
// Manual 2x k-unroll with two named fragment sets: set1 (k+32) loads while
// set0 (k) WMMAs issue and vice versa -- double buffering with zero register
// moves (requires K % 64 == 0, true for all uses here: K = 2048).
// ---------------------------------------------------------------------------
static __device__ inline void gemm_mainloop(const __bf16* __restrict__ A,
                                            const __bf16* __restrict__ W,
                                            int K, int rowbase, int colbase,
                                            int lane, v8f acc[4][2]) {
  int m = lane & 15;
  v16bf a0[4], b0[2], a1[4], b1[2];
#pragma unroll
  for (int i = 0; i < 4; ++i) a0[i] = load_afrag(A, K, rowbase + i * 16 + m, 0, lane);
#pragma unroll
  for (int j = 0; j < 2; ++j) b0[j] = load_bfrag(W, K, colbase + j * 16 + m, 0, lane);

  for (int k = 0; k < K; k += 64) {
    // fragments for k+32 (always in range: K % 64 == 0)
#pragma unroll
    for (int i = 0; i < 4; ++i) a1[i] = load_afrag(A, K, rowbase + i * 16 + m, k + 32, lane);
#pragma unroll
    for (int j = 0; j < 2; ++j) b1[j] = load_bfrag(W, K, colbase + j * 16 + m, k + 32, lane);
    if (k + 96 < K) {  // near-scope prefetch ahead of the load stream
      __builtin_prefetch(A + (size_t)(rowbase + m) * K + k + 96, 0, 3);
      __builtin_prefetch(W + (size_t)(colbase + m) * K + k + 96, 0, 3);
    }
#pragma unroll
    for (int i = 0; i < 4; ++i)
#pragma unroll
      for (int j = 0; j < 2; ++j)
        acc[i][j] = wmma_bf16(a0[i], b0[j], acc[i][j]);

    if (k + 64 < K) {  // fragments for k+64 overwrite set0 after its WMMAs
#pragma unroll
      for (int i = 0; i < 4; ++i) a0[i] = load_afrag(A, K, rowbase + i * 16 + m, k + 64, lane);
#pragma unroll
      for (int j = 0; j < 2; ++j) b0[j] = load_bfrag(W, K, colbase + j * 16 + m, k + 64, lane);
    }
#pragma unroll
    for (int i = 0; i < 4; ++i)
#pragma unroll
      for (int j = 0; j < 2; ++j)
        acc[i][j] = wmma_bf16(a1[i], b1[j], acc[i][j]);
  }
}

// QKV projection: bias add, write bf16 head-major [B, Hh, S, D]
__global__ __launch_bounds__(256) void gemm_bias_headmajor(
    const __bf16* __restrict__ A, const __bf16* __restrict__ W,
    const float* __restrict__ bias, __bf16* __restrict__ out, int K, int Hh) {
  int lane = threadIdx.x & 31, w = threadIdx.x >> 5;
  int rowbase = blockIdx.y * 128 + (w >> 2) * 64;
  int colbase = blockIdx.x * 128 + (w & 3) * 32;
  v8f acc[4][2];
#pragma unroll
  for (int i = 0; i < 4; ++i)
#pragma unroll
    for (int j = 0; j < 2; ++j) acc[i][j] = vzero8();

  gemm_mainloop(A, W, K, rowbase, colbase, lane, acc);

  int n = lane & 15, g = lane >> 4;
#pragma unroll
  for (int i = 0; i < 4; ++i)
#pragma unroll
    for (int j = 0; j < 2; ++j)
#pragma unroll
      for (int r = 0; r < 8; ++r) {
        int row = rowbase + i * 16 + r + g * 8;   // C/D layout: vgpr r, lane half g
        int col = colbase + j * 16 + n;
        float val = acc[i][j][r] + bias[col];
        int b = row / S_SZ, s = row % S_SZ;
        int h = col / D_HD, d = col % D_HD;
        out[(((size_t)b * Hh + h) * S_SZ + s) * D_HD + d] = (__bf16)val;
      }
}

// Output projection: bias add, write fp32 row-major [M, N]
__global__ __launch_bounds__(256) void gemm_bias_f32out(
    const __bf16* __restrict__ A, const __bf16* __restrict__ W,
    const float* __restrict__ bias, float* __restrict__ out, int K, int N) {
  int lane = threadIdx.x & 31, w = threadIdx.x >> 5;
  int rowbase = blockIdx.y * 128 + (w >> 2) * 64;
  int colbase = blockIdx.x * 128 + (w & 3) * 32;
  v8f acc[4][2];
#pragma unroll
  for (int i = 0; i < 4; ++i)
#pragma unroll
    for (int j = 0; j < 2; ++j) acc[i][j] = vzero8();

  gemm_mainloop(A, W, K, rowbase, colbase, lane, acc);

  int n = lane & 15, g = lane >> 4;
#pragma unroll
  for (int i = 0; i < 4; ++i)
#pragma unroll
    for (int j = 0; j < 2; ++j)
#pragma unroll
      for (int r = 0; r < 8; ++r) {
        int row = rowbase + i * 16 + r + g * 8;
        int col = colbase + j * 16 + n;
        out[(size_t)row * N + col] = acc[i][j][r] + bias[col];
      }
}

// ---------------------------------------------------------------------------
// RoPE in place on bf16 head-major tensors. One thread per (row, freq-pair).
// ---------------------------------------------------------------------------
__global__ void rope_bf16(__bf16* __restrict__ x, long nrows) {
  long idx = (long)blockIdx.x * blockDim.x + threadIdx.x;
  long total = nrows * (D_HD / 2);
  if (idx >= total) return;
  int  i   = (int)(idx & 63);
  long row = idx >> 6;
  int  s   = (int)(row % S_SZ);
  // inv_freq = theta^-(2i/D); ln(10000) = 9.210340371976184
  float freq = (float)s * __expf(-((float)(2 * i) / (float)D_HD) * 9.210340371976184f);
  float c = __cosf(freq), sn = __sinf(freq);
  __bf16* p = x + row * D_HD;
  float x1 = (float)p[i], x2 = (float)p[i + 64];
  p[i]      = (__bf16)(x1 * c - x2 * sn);
  p[i + 64] = (__bf16)(x2 * c + x1 * sn);
}

// ---------------------------------------------------------------------------
// Flash attention (causal, GQA). Block = 8 waves; each wave owns a 16-row
// query tile; waves share LDS-staged K/V for their (b,h). K tile padded to
// 136 elems/row (272 B: lane-bank stride 4 -> conflict-free B-frag reads).
// V is transposed once per block into sVT[d][key] (40 elems/row, bank stride
// 20 -> all 16 lanes distinct) so P*V B-fragments are two ds_load_b128s.
// ---------------------------------------------------------------------------
__global__ __launch_bounds__(256) void flash_attn_kernel(
    const __bf16* __restrict__ Qh, const __bf16* __restrict__ Kh,
    const __bf16* __restrict__ Vh, __bf16* __restrict__ attn) {
  __shared__ __bf16 sK[32][136];     // keys x D, padded
  __shared__ __bf16 sV[32][136];     // keys x D, padded (transpose source)
  __shared__ __bf16 sVT[128][40];    // D x keys, padded
  __shared__ __bf16 sP[8][16][40];   // per-wave prob tile (D-layout -> A-layout relay)

  const float scale = 0.08838834764831845f;  // D^-0.5

  int bh = blockIdx.y;
  int b = bh / H_Q, h = bh % H_Q;
  int hk = h / (H_Q / H_KV);
  int qbase = blockIdx.x * 128;
  int w = threadIdx.x >> 5, lane = threadIdx.x & 31;
  int n = lane & 15, g = lane >> 4;
  int qrow0 = qbase + w * 16;

  const __bf16* Qp = Qh + ((size_t)b * H_Q + h) * S_SZ * D_HD;
  const __bf16* Kp = Kh + ((size_t)b * H_KV + hk) * S_SZ * D_HD;
  const __bf16* Vp = Vh + ((size_t)b * H_KV + hk) * S_SZ * D_HD;

  // Whole 16x128 Q tile lives in registers as 4 A-fragments (K chunks of 32).
  v16bf qf[4];
#pragma unroll
  for (int kk = 0; kk < 4; ++kk)
    qf[kk] = load_afrag(Qp, D_HD, qrow0 + n, kk * 32, lane);

  v8f oacc[8];
#pragma unroll
  for (int dt = 0; dt < 8; ++dt) oacc[dt] = vzero8();
  float mst[8], lst[8];
#pragma unroll
  for (int r = 0; r < 8; ++r) { mst[r] = -1e30f; lst[r] = 0.0f; }

  int kend = qbase + 128;  // causal: no keys beyond the block's last query row
  for (int kb = 0; kb < kend; kb += 32) {
    __syncthreads();  // protect previous iteration's LDS reads
    {   // stage K and V row-major (coalesced global, contiguous ds stores)
      int t = threadIdx.x;
      int r = t >> 3;            // 0..31 key rows
      int c = (t & 7) * 16;      // 16 bf16 (32 B) per thread per tensor
      const uint4* gk = (const uint4*)(Kp + (size_t)(kb + r) * D_HD + c);
      uint4* dk = (uint4*)&sK[r][c];
      dk[0] = gk[0]; dk[1] = gk[1];
      const uint4* gv = (const uint4*)(Vp + (size_t)(kb + r) * D_HD + c);
      uint4* dv = (uint4*)&sV[r][c];
      dv[0] = gv[0]; dv[1] = gv[1];
    }
    __syncthreads();
    {   // cooperative transpose V -> sVT (amortized over all 8 waves)
      int t = threadIdx.x;
      int d = t >> 1;            // 0..127 head dims
      int k0 = (t & 1) * 16;     // key half
      FragU tmp;
#pragma unroll
      for (int j = 0; j < 16; ++j) tmp.b[j] = sV[k0 + j][d];
      uint4* dst = (uint4*)&sVT[d][k0];
      dst[0] = tmp.u[0];
      dst[1] = tmp.u[1];
    }
    __syncthreads();

    // wave-uniform causal skip: this wave needs keys only up to qrow0+15
    if (kb <= qrow0 + 15) {
      // ---- scores: two 16-key subtiles, S = Q (16xD) x K^T (Dx16) ----
      v8f sc[2];
#pragma unroll
      for (int sub = 0; sub < 2; ++sub) {
        v8f s = vzero8();
#pragma unroll
        for (int kk = 0; kk < 4; ++kk) {
          // B-frag column n = key (sub*16+n); 32 contiguous bytes of that key's dims
          const __bf16* p = &sK[sub * 16 + n][kk * 32 + g * 16];
          FragU f;
          f.u[0] = *(const uint4*)(p);
          f.u[1] = *(const uint4*)(p + 8);
          s = wmma_bf16(qf[kk], f.v, s);
        }
        int colk = kb + sub * 16 + n;
#pragma unroll
        for (int r = 0; r < 8; ++r) {
          int row = qrow0 + r + g * 8;
          float v = s[r] * scale;
          if (colk > row) v = -1e30f;  // causal mask
          s[r] = v;
        }
        sc[sub] = s;
      }

      // ---- online softmax (row stats replicated across each 16-lane group) ----
      float corr[8];
#pragma unroll
      for (int r = 0; r < 8; ++r) {
        float rm = fmaxf(sc[0][r], sc[1][r]);
#pragma unroll
        for (int off = 1; off < 16; off <<= 1)
          rm = fmaxf(rm, __shfl_xor(rm, off, 32));
        float nm = fmaxf(mst[r], rm);
        float cr = __expf(mst[r] - nm);
        mst[r] = nm; corr[r] = cr;
        float p0 = __expf(sc[0][r] - nm);
        float p1 = __expf(sc[1][r] - nm);
        float ps = p0 + p1;
#pragma unroll
        for (int off = 1; off < 16; off <<= 1)
          ps += __shfl_xor(ps, off, 32);
        lst[r] = lst[r] * cr + ps;
        // stash probs (D-layout coordinates) into per-wave LDS scratch
        sP[w][r + g * 8][n]      = (__bf16)p0;
        sP[w][r + g * 8][16 + n] = (__bf16)p1;
      }

      // Re-read probs in A-layout (same-wave LDS ops are in-order; no barrier)
      const __bf16* pp = &sP[w][n][g * 16];
      FragU pf;
      pf.u[0] = *(const uint4*)(pp);
      pf.u[1] = *(const uint4*)(pp + 8);

      // ---- O += P (16x32) x V (32xD), with rescale of old accumulator ----
#pragma unroll
      for (int dt = 0; dt < 8; ++dt) {
#pragma unroll
        for (int r = 0; r < 8; ++r) oacc[dt][r] *= corr[r];
        // B-frag column = head dim dt*16+n; keys contiguous in sVT
        const __bf16* vp = &sVT[dt * 16 + n][g * 16];
        FragU vf;
        vf.u[0] = *(const uint4*)(vp);
        vf.u[1] = *(const uint4*)(vp + 8);
        oacc[dt] = wmma_bf16(pf.v, vf.v, oacc[dt]);
      }
    }
  }

  // ---- epilogue: normalize and write bf16 [B, S, E] ----
#pragma unroll
  for (int dt = 0; dt < 8; ++dt)
#pragma unroll
    for (int r = 0; r < 8; ++r) {
      int row = qrow0 + r + g * 8;
      int col = h * D_HD + dt * 16 + n;
      attn[((size_t)b * S_SZ + row) * E_SZ + col] = (__bf16)(oacc[dt][r] / lst[r]);
    }
}

// ---------------------------------------------------------------------------
// Launch
// ---------------------------------------------------------------------------
extern "C" void kernel_launch(void* const* d_in, const int* in_sizes, int n_in,
                              void* d_out, int out_size, void* d_ws, size_t ws_size,
                              hipStream_t stream) {
  (void)in_sizes; (void)n_in; (void)out_size; (void)ws_size;

  const float* x  = (const float*)d_in[0];
  const float* wq = (const float*)d_in[1];
  const float* bq = (const float*)d_in[2];
  const float* wk = (const float*)d_in[3];
  const float* bk = (const float*)d_in[4];
  const float* wv = (const float*)d_in[5];
  const float* bv = (const float*)d_in[6];
  const float* wo = (const float*)d_in[7];
  const float* bo = (const float*)d_in[8];
  float* out = (float*)d_out;

  char* ws = (char*)d_ws;
  size_t off = 0;
  auto alloc = [&](size_t bytes) -> void* {
    void* p = ws + off;
    off += (bytes + 255) & ~(size_t)255;
    return p;
  };
  __bf16* xb    = (__bf16*)alloc((size_t)M_ROWS * E_SZ * 2);   // 16 MiB
  __bf16* wqb   = (__bf16*)alloc((size_t)E_SZ * E_SZ * 2);     //  8 MiB
  __bf16* wkb   = (__bf16*)alloc((size_t)KV_SZ * E_SZ * 2);    //  2 MiB
  __bf16* wvb   = (__bf16*)alloc((size_t)KV_SZ * E_SZ * 2);    //  2 MiB
  __bf16* wob   = (__bf16*)alloc((size_t)E_SZ * E_SZ * 2);     //  8 MiB
  __bf16* qh    = (__bf16*)alloc((size_t)B_SZ * H_Q * S_SZ * D_HD * 2);   // 16 MiB
  __bf16* kh    = (__bf16*)alloc((size_t)B_SZ * H_KV * S_SZ * D_HD * 2);  //  4 MiB
  __bf16* vh    = (__bf16*)alloc((size_t)B_SZ * H_KV * S_SZ * D_HD * 2);  //  4 MiB
  __bf16* attnb = (__bf16*)alloc((size_t)M_ROWS * E_SZ * 2);   // 16 MiB

  // 1) convert to bf16
  cvt_bf16_kernel<<<2048, 256, 0, stream>>>(x,  xb,  (long)M_ROWS * E_SZ);
  cvt_bf16_kernel<<<2048, 256, 0, stream>>>(wq, wqb, (long)E_SZ * E_SZ);
  cvt_bf16_kernel<<<1024, 256, 0, stream>>>(wk, wkb, (long)KV_SZ * E_SZ);
  cvt_bf16_kernel<<<1024, 256, 0, stream>>>(wv, wvb, (long)KV_SZ * E_SZ);
  cvt_bf16_kernel<<<2048, 256, 0, stream>>>(wo, wob, (long)E_SZ * E_SZ);

  // 2) QKV projections (WMMA), bf16 head-major output
  gemm_bias_headmajor<<<dim3(E_SZ / 128,  M_ROWS / 128), 256, 0, stream>>>(xb, wqb, bq, qh, E_SZ, H_Q);
  gemm_bias_headmajor<<<dim3(KV_SZ / 128, M_ROWS / 128), 256, 0, stream>>>(xb, wkb, bk, kh, E_SZ, H_KV);
  gemm_bias_headmajor<<<dim3(KV_SZ / 128, M_ROWS / 128), 256, 0, stream>>>(xb, wvb, bv, vh, E_SZ, H_KV);

  // 3) RoPE in place on Q and K
  {
    long nq = (long)B_SZ * H_Q * S_SZ;
    long nk = (long)B_SZ * H_KV * S_SZ;
    rope_bf16<<<(unsigned)((nq * 64 + 255) / 256), 256, 0, stream>>>(qh, nq);
    rope_bf16<<<(unsigned)((nk * 64 + 255) / 256), 256, 0, stream>>>(kh, nk);
  }

  // 4) causal flash attention (WMMA), bf16 [B,S,E] output
  flash_attn_kernel<<<dim3(S_SZ / 128, B_SZ * H_Q), 256, 0, stream>>>(qh, kh, vh, attnb);

  // 5) output projection (WMMA), fp32 to d_out
  gemm_bias_f32out<<<dim3(E_SZ / 128, M_ROWS / 128), 256, 0, stream>>>(attnb, wob, bo, out, E_SZ, E_SZ);
}
